// DPFL_326417514722
// MI455X (gfx1250) — matrix operand order repaired and checked by reading, not compiled, not run on hardware
//
#include <hip/hip_runtime.h>
#include <hip/hip_bf16.h>
#include <utility>

// ---------------------------------------------------------------------------
// Model constants (match reference)
// ---------------------------------------------------------------------------
#define NB   16      // batch N
#define TT   2048    // time T
#define NHEAD 4
#define HD   16      // NF / NHEAD

typedef __attribute__((ext_vector_type(16))) _Float16 v16h;
typedef __attribute__((ext_vector_type(8)))  _Float16 h8v;
typedef __attribute__((ext_vector_type(4)))  _Float16 h4v;
typedef __attribute__((ext_vector_type(8)))  float    v8f;

// HEAD(3), HAND(12), HIP(4), FOOT(6), ALL(25)
__constant__ int g_nodes[50] = {
  2,3,20,
  4,5,6,7,8,9,10,11,21,22,23,24,
  0,1,12,16,
  13,14,15,17,18,19,
  0,1,2,3,4,5,6,7,8,9,10,11,12,13,14,15,16,17,18,19,20,21,22,23,24
};

__device__ __forceinline__ h4v cvt4(float4 f) {
  h4v h; h[0] = (_Float16)f.x; h[1] = (_Float16)f.y; h[2] = (_Float16)f.z; h[3] = (_Float16)f.w;
  return h;
}

// Async global->LDS copy of 16 bytes (CDNA5, ASYNCcnt-tracked), ISA 08 §4.
// ldsOff = LDS byte offset (low 32 bits of generic shared pointer).
__device__ __forceinline__ void async_copy_b128(unsigned ldsOff, const float* gsrc) {
  unsigned long long ga = (unsigned long long)gsrc;
  asm volatile("global_load_async_to_lds_b128 %0, %1, off"
               :: "v"(ldsOff), "v"(ga) : "memory");
}
__device__ __forceinline__ void async_wait0() {
  asm volatile("s_wait_asynccnt 0x0" ::: "memory");
}

// ---------------------------------------------------------------------------
// WMMA fragment helpers (contiguous k = 16*(lane/16) + s order for A and B, wave32)
// ldsA layout: [m][k] (64 x 32).  Packed weights: [col][K] f16 in global.
// ---------------------------------------------------------------------------
__device__ __forceinline__ v16h load_a_frag(const _Float16* ldsA, int m0, int lane) {
  int m  = m0 + (lane & 15);
  int kb = (lane < 16) ? 0 : 16;
  const _Float16* p = ldsA + m * 32 + kb;
  h8v lo = *(const h8v*)(p);        // K = kb .. kb+7
  h8v hi = *(const h8v*)(p + 8);    // K = kb+8 .. kb+15
  v16h a;
#pragma unroll
  for (int i = 0; i < 8; ++i) { a[i] = lo[i]; a[8 + i] = hi[i]; }
  return a;
}

// B fragment straight from packed f16 global weight [col][K] (L2-resident)
__device__ __forceinline__ v16h load_b_packed(const _Float16* Wp, int K, int k0, int n0, int lane) {
  int col = n0 + (lane & 15);
  int kb  = (lane < 16) ? 0 : 16;
  const _Float16* p = Wp + (long)col * K + k0 + kb;
  h8v lo = *(const h8v*)(p);
  h8v hi = *(const h8v*)(p + 8);
  v16h b;
#pragma unroll
  for (int i = 0; i < 8; ++i) { b[i] = lo[i]; b[8 + i] = hi[i]; }
  return b;
}

template<int ACT> __device__ __forceinline__ float act_fn(float x) {
  if (ACT == 1) return fmaxf(x, 0.f);
  if (ACT == 2) return 1.f / (1.f + __expf(-x));
  if (ACT == 3) return 0.5f * x * (1.f + erff(x * 0.70710678118654752f)); // exact gelu
  return x;
}

// Coalesced epilogue through LDS: acc tiles -> ldsO -> vectorized global.
// post order: +bias, ACT, (*mask if MPRE), (+res if RES), (*mask if MPOST)
template<int ACT, bool MPRE, bool RES, bool MPOST>
__device__ __forceinline__ void epilogue64(float* ldsO, v8f acc[4], int m0, int lane, int tid,
                                           long rowBase, const float* bias, const float* res,
                                           const float* mask, float* Y) {
#pragma unroll
  for (int nt = 0; nt < 4; ++nt) {
    int col = nt * 16 + (lane & 15);
#pragma unroll
    for (int r = 0; r < 8; ++r) {
      int mr = m0 + ((lane < 16) ? r : 8 + r);
      ldsO[mr * 64 + col] = acc[nt][r];
    }
  }
  __syncthreads();
  int rr = tid >> 1, cb = (tid & 1) * 32;        // 32 contiguous floats / thread
  long row = rowBase + rr;
  float mv = 1.f;
  if (MPRE || MPOST) mv = mask[row];
#pragma unroll
  for (int u = 0; u < 8; ++u) {
    float4 v = *(float4*)&ldsO[rr * 64 + cb + u * 4];
    float4 bv = make_float4(0.f, 0.f, 0.f, 0.f);
    if (bias) bv = *(const float4*)&bias[cb + u * 4];
    float4 rv = make_float4(0.f, 0.f, 0.f, 0.f);
    if (RES) rv = *(const float4*)&res[row * 64 + cb + u * 4];
#define EPI1(X_, B_, R_) { float t_ = (X_) + (B_); t_ = act_fn<ACT>(t_); \
    if (MPRE) t_ *= mv; if (RES) t_ += (R_); if (MPOST) t_ *= mv; (X_) = t_; }
    EPI1(v.x, bv.x, rv.x) EPI1(v.y, bv.y, rv.y) EPI1(v.z, bv.z, rv.z) EPI1(v.w, bv.w, rv.w)
#undef EPI1
    *(float4*)&Y[row * 64 + cb + u * 4] = v;
  }
}

// ---------------------------------------------------------------------------
// Weight pre-pack: fp32 -> f16 [col][K].
// mode 0: W stored (K,64) in-major  -> P[c*K+k] = W[k*64+c]
// mode 1: W stored (64,K) out-major / flat copy -> P[i] = W[i]
// mode 2: ffw (64,64,3) -> P[o*192 + seg*64 + cc] = W[(o*64+cc)*3 + seg]
// ---------------------------------------------------------------------------
__global__ void pack_kernel(const float* __restrict__ W, _Float16* __restrict__ P,
                            int K, int total, int mode) {
  int i = blockIdx.x * 256 + threadIdx.x;
  if (i >= total) return;
  float v;
  if (mode == 0) {
    int c = i / K, k = i - c * K;
    v = W[k * 64 + c];
  } else if (mode == 2) {
    int o = i / 192, k = i - o * 192;
    int seg = k >> 6, cc = k & 63;
    v = W[(o * 64 + cc) * 3 + seg];
  } else {
    v = W[i];
  }
  P[i] = (_Float16)v;
}

// ---------------------------------------------------------------------------
// Generic WMMA GEMM:  Y[row, 0..63] = post( X[row, 0..K-1] @ Wpacked + b )
// rows = NB*TT (grid.x = 512, 64 rows/block), K multiple of 32.
// wStride != 0 -> per-sample packed weight (W + n*wStride halves), n = row/TT.
// ---------------------------------------------------------------------------
template<int ACT, bool MPRE, bool RES, bool MPOST>
__global__ void gemm64_kernel(const float* __restrict__ X, const _Float16* __restrict__ W,
                              int wStride, const float* __restrict__ bias,
                              const float* __restrict__ res, const float* __restrict__ mask,
                              float* __restrict__ Y, int K) {
  __shared__ __align__(32) float smemf[64 * 64];        // 16KB; low 4KB = f16 A tile
  _Float16* ldsA = (_Float16*)smemf;
  const int rowBase = blockIdx.x * 64;
  const int n = rowBase >> 11;                 // TT = 2048
  const _Float16* Wp = W + (long)n * wStride;
  const int tid = threadIdx.x;                 // 128 threads, 4 waves
  const int lane = tid & 31;
  const int m0 = (tid >> 5) * 16;
  const int e0 = tid * 16;

  v8f acc[4];
#pragma unroll
  for (int i = 0; i < 4; ++i)
#pragma unroll
    for (int e = 0; e < 8; ++e) acc[i][e] = 0.f;

  for (int k0 = 0; k0 < K; k0 += 32) {
    { // stage A: thread -> row r, 16 consecutive k (vector both sides)
      int r = e0 >> 5, kb = e0 & 31;
      const float* src = &X[(long)(rowBase + r) * K + k0 + kb];
#pragma unroll
      for (int u = 0; u < 4; ++u)
        *(h4v*)&ldsA[e0 + u * 4] = cvt4(*(const float4*)(src + u * 4));
    }
    __syncthreads();
    v16h a = load_a_frag(ldsA, m0, lane);
#pragma unroll
    for (int nt = 0; nt < 4; ++nt) {
      v16h b = load_b_packed(Wp, K, k0, nt * 16, lane);
      acc[nt] = __builtin_amdgcn_wmma_f32_16x16x32_f16(
          false, a, false, b, (short)0, acc[nt], false, false);
    }
    __syncthreads();
  }
  epilogue64<ACT, MPRE, RES, MPOST>(smemf, acc, m0, lane, tid, rowBase, bias, res, mask, Y);
}

// ---------------------------------------------------------------------------
// Dilated conv3 (K=192 implicit im2col, packed ffw) + ReLU. grid (TT/64, NB).
// Each 32-wide K step lies entirely inside one seg -> uniform bounds check.
// ---------------------------------------------------------------------------
__global__ void dilconv_kernel(const float* __restrict__ X, const _Float16* __restrict__ Wp,
                               const float* __restrict__ ffb, float* __restrict__ Y, int dil) {
  __shared__ __align__(32) float smemf[64 * 64];
  _Float16* ldsA = (_Float16*)smemf;
  const int n = blockIdx.y;
  const int t0 = blockIdx.x * 64;
  const int tid = threadIdx.x, lane = tid & 31, m0 = (tid >> 5) * 16;
  const int e0 = tid * 16;

  v8f acc[4];
#pragma unroll
  for (int i = 0; i < 4; ++i)
#pragma unroll
    for (int e = 0; e < 8; ++e) acc[i][e] = 0.f;

  for (int k0 = 0; k0 < 192; k0 += 32) {
    const int seg = k0 >> 6;                  // uniform per step
    const int cb  = k0 & 63;                  // 0 or 32
    { // stage A: row r -> time t0+r shifted by (seg-1)*dil; 16 consecutive c
      int r = e0 >> 5, kb = e0 & 31;
      int ts = t0 + r + (seg - 1) * dil;
      float4 f[4];
#pragma unroll
      for (int u = 0; u < 4; ++u) f[u] = make_float4(0.f, 0.f, 0.f, 0.f);
      if (ts >= 0 && ts < TT) {
        const float* src = &X[((long)(n * TT + ts)) * 64 + cb + kb];
#pragma unroll
        for (int u = 0; u < 4; ++u) f[u] = *(const float4*)(src + u * 4);
      }
#pragma unroll
      for (int u = 0; u < 4; ++u) *(h4v*)&ldsA[e0 + u * 4] = cvt4(f[u]);
    }
    __syncthreads();
    v16h a = load_a_frag(ldsA, m0, lane);
#pragma unroll
    for (int nt = 0; nt < 4; ++nt) {
      v16h b = load_b_packed(Wp, 192, k0, nt * 16, lane);
      acc[nt] = __builtin_amdgcn_wmma_f32_16x16x32_f16(
          false, a, false, b, (short)0, acc[nt], false, false);
    }
    __syncthreads();
  }
  epilogue64<1, false, false, false>(smemf, acc, m0, lane, tid, (long)(n * TT + t0),
                                     ffb, nullptr, nullptr, Y);
}

// ---------------------------------------------------------------------------
// Correlation: AMAP[n,c,d] = (1/T) sum_t FS[n,t,c]*FT[n,t,d]   (K = 2048 WMMA)
// grid NB blocks, 128 threads (true transpose: strided gathers, vector LDS).
// ---------------------------------------------------------------------------
__global__ void corr_kernel(const float* __restrict__ FS, const float* __restrict__ FT,
                            float* __restrict__ AMAP) {
  __shared__ __align__(32) _Float16 ldsA[64 * 32];   // [c][kk] = FS[t0+kk, c]
  __shared__ __align__(32) _Float16 ldsB[64 * 32];   // [d][kk] = FT[t0+kk, d]
  const int n = blockIdx.x;
  const int tid = threadIdx.x, lane = tid & 31, m0 = (tid >> 5) * 16;
  const int e0 = tid * 16;
  const int cch = e0 >> 5, kb = e0 & 31;

  v8f acc[4];
#pragma unroll
  for (int i = 0; i < 4; ++i)
#pragma unroll
    for (int e = 0; e < 8; ++e) acc[i][e] = 0.f;

  for (int t0 = 0; t0 < TT; t0 += 32) {
    const float* fsrc = &FS[((long)(n * TT + t0 + kb)) * 64 + cch];
    const float* tsrc = &FT[((long)(n * TT + t0 + kb)) * 64 + cch];
#pragma unroll
    for (int u = 0; u < 4; ++u) {
      h4v ha, hb;
#pragma unroll
      for (int q = 0; q < 4; ++q) {
        ha[q] = (_Float16)fsrc[(u * 4 + q) * 64];
        hb[q] = (_Float16)tsrc[(u * 4 + q) * 64];
      }
      *(h4v*)&ldsA[e0 + u * 4] = ha;
      *(h4v*)&ldsB[e0 + u * 4] = hb;
    }
    __syncthreads();
    v16h a = load_a_frag(ldsA, m0, lane);
#pragma unroll
    for (int nt = 0; nt < 4; ++nt) {
      int col = nt * 16 + (lane & 15);
      int kb2 = (lane < 16) ? 0 : 16;
      const _Float16* p = ldsB + col * 32 + kb2;
      h8v lo = *(const h8v*)(p);
      h8v hi = *(const h8v*)(p + 8);
      v16h b;
#pragma unroll
      for (int i = 0; i < 8; ++i) { b[i] = lo[i]; b[8 + i] = hi[i]; }
      acc[nt] = __builtin_amdgcn_wmma_f32_16x16x32_f16(
          false, a, false, b, (short)0, acc[nt], false, false);
    }
    __syncthreads();
  }
#pragma unroll
  for (int nt = 0; nt < 4; ++nt) {
    int d = nt * 16 + (lane & 15);
#pragma unroll
    for (int r = 0; r < 8; ++r) {
      int c = m0 + ((lane < 16) ? r : 8 + r);
      AMAP[(long)n * 4096 + c * 64 + d] = acc[nt][r] * (1.f / (float)TT);
    }
  }
}

// ---------------------------------------------------------------------------
// small VALU kernels
// ---------------------------------------------------------------------------
__global__ void softmax64_kernel(float* __restrict__ A) {          // grid NB, 64 thr
  float* row = A + (long)blockIdx.x * 4096 + threadIdx.x * 64;
  float mx = -1e30f;
  for (int d = 0; d < 64; ++d) mx = fmaxf(mx, row[d]);
  float s = 0.f;
  for (int d = 0; d < 64; ++d) { float e = __expf(row[d] - mx); row[d] = e; s += e; }
  float inv = 1.f / s;
  for (int d = 0; d < 64; ++d) row[d] *= inv;
}

__global__ void instnorm_stats_kernel(const float* __restrict__ X, float* __restrict__ st) {
  // grid NB*64 (n*64+c), 128 threads
  int n = blockIdx.x >> 6, c = blockIdx.x & 63;
  __shared__ float s1[128], s2[128];
  float a = 0.f, b = 0.f;
  for (int t = threadIdx.x; t < TT; t += 128) {
    float v = X[((long)(n * TT + t)) * 64 + c];
    a += v; b += v * v;
  }
  s1[threadIdx.x] = a; s2[threadIdx.x] = b; __syncthreads();
  for (int s = 64; s > 0; s >>= 1) {
    if (threadIdx.x < s) { s1[threadIdx.x] += s1[threadIdx.x + s]; s2[threadIdx.x] += s2[threadIdx.x + s]; }
    __syncthreads();
  }
  if (threadIdx.x == 0) {
    float m = s1[0] / (float)TT;
    float var = s2[0] / (float)TT - m * m;
    st[blockIdx.x * 2] = m;
    st[blockIdx.x * 2 + 1] = rsqrtf(var + 1e-5f);
  }
}

__global__ void instnorm_apply_kernel(const float* __restrict__ X, const float* __restrict__ st,
                                      float* __restrict__ Y) {     // grid 8192, 256 thr
  long i = (long)blockIdx.x * 256 + threadIdx.x;
  int c = i & 63;
  int n = (int)(i >> 17);                                          // / (2048*64)
  float m = st[(n * 64 + c) * 2], r = st[(n * 64 + c) * 2 + 1];
  Y[i] = (X[i] - m) * r;
}

__global__ void kv_kernel(const float* __restrict__ Kb, const float* __restrict__ Vb,
                          float* __restrict__ KV, float* __restrict__ Ks) {
  // grid (NB, NHEAD), 256 threads.  KV[d,e] = sum_l K[l,d]V[l,e]; Ks[d]=sum_l K + 1e-6
  // Tile staging uses CDNA5 async global->LDS copies (pure fp32 copy).
  int n = blockIdx.x, h = blockIdx.y;
  __shared__ __align__(16) float lk[64 * 16], lv[64 * 16];
  int tid = threadIdx.x;
  int d = tid >> 4, e = tid & 15;
  int l0 = tid >> 2, q0 = (tid & 3) * 4;            // this thread's 16B chunk
  unsigned lkoff = (unsigned)(unsigned long long)&lk[tid * 4];
  unsigned lvoff = (unsigned)(unsigned long long)&lv[tid * 4];
  float acc = 0.f, ks = 0.f;
  for (int t0 = 0; t0 < TT; t0 += 64) {
    long gbase = ((long)(n * TT + t0 + l0)) * 64 + h * HD + q0;
    async_copy_b128(lkoff, Kb + gbase);
    async_copy_b128(lvoff, Vb + gbase);
    async_wait0();
    __syncthreads();
    for (int l = 0; l < 64; ++l) acc += lk[l * 16 + d] * lv[l * 16 + e];
    if (e == 0) for (int l = 0; l < 64; ++l) ks += lk[l * 16 + d];
    __syncthreads();
  }
  KV[(((long)n * NHEAD + h) * 16 + d) * 16 + e] = acc;
  if (e == 0) Ks[((long)n * NHEAD + h) * 16 + d] = ks + 1e-6f;
}

__global__ void attapply_kernel(const float* __restrict__ Qb, const float* __restrict__ KV,
                                const float* __restrict__ Ks, float* __restrict__ O) {
  // grid (NB, TT/64), 256 thr: l_local = tid>>2, h = tid&3
  int n = blockIdx.x, t0 = blockIdx.y * 64;
  __shared__ __align__(16) float lkv[NHEAD * 256];
  __shared__ float lks[NHEAD * 16];
  int tid = threadIdx.x;
  // async preload of the per-sample KV block (1024 floats, 4 per thread)
  unsigned lkvoff = (unsigned)(unsigned long long)&lkv[tid * 4];
  async_copy_b128(lkvoff, KV + (long)n * NHEAD * 256 + tid * 4);
  if (tid < NHEAD * 16) lks[tid] = Ks[(long)n * NHEAD * 16 + tid];
  async_wait0();
  __syncthreads();
  int ll = tid >> 2, h = tid & 3;
  long base = ((long)(n * TT + t0 + ll)) * 64 + h * HD;
  float q[16]; float den = 0.f;
#pragma unroll
  for (int d = 0; d < 16; ++d) { q[d] = Qb[base + d]; den += q[d] * lks[h * 16 + d]; }
  float z = 1.f / den;
#pragma unroll
  for (int e = 0; e < 16; ++e) {
    float s = 0.f;
#pragma unroll
    for (int d = 0; d < 16; ++d) s += q[d] * lkv[(h * 16 + d) * 16 + e];
    O[base + e] = z * s;
  }
}

__global__ void conv_in_kernel(const float* __restrict__ x, const float* __restrict__ w,
                               const float* __restrict__ b, const float* __restrict__ mask,
                               float* __restrict__ x10) {
  long i = (long)blockIdx.x * 256 + threadIdx.x;   // N*10*T*25
  if (i >= (long)NB * 10 * TT * 25) return;
  int v = (int)(i % 25); long r = i / 25;
  int t = (int)(r % TT); long r2 = r / TT;
  int o = (int)(r2 % 10); int n = (int)(r2 / 10);
  float s = b[o];
  for (int c = 0; c < 3; ++c)
    s += x[(((long)(n * 3 + c)) * TT + t) * 25 + v] * w[o * 3 + c];
  x10[i] = s * mask[n * TT + t];
}

__global__ void nodeconv_kernel(const float* __restrict__ x10, const float* __restrict__ w,
                                const float* __restrict__ b, const float* __restrict__ mask,
                                float* __restrict__ Y, int ch, int listOff, int nlen, int applyMask) {
  long i = (long)blockIdx.x * 256 + threadIdx.x;   // N*T*64
  int o = i & 63; long nt = i >> 6;
  int t = (int)(nt & (TT - 1)); int n = (int)(nt >> 11);
  const float* xp = x10 + (((long)(n * 10 + ch)) * TT + t) * 25;
  float s = b[o];
  for (int j = 0; j < nlen; ++j) s += xp[g_nodes[listOff + j]] * w[o * nlen + j];
  if (applyMask) s *= mask[nt];
  Y[i] = s;
}

__global__ void concat_kernel(const float* f0, const float* f1, const float* f2,
                              const float* f3, const float* f4,
                              const float* __restrict__ mask, float* __restrict__ fuse) {
  long i = (long)blockIdx.x * 256 + threadIdx.x;   // N*T*320
  int c = (int)(i % 320); long nt = i / 320;
  int part = c >> 6, cc = c & 63;
  const float* f = (part == 0) ? f0 : (part == 1) ? f1 : (part == 2) ? f2 : (part == 3) ? f3 : f4;
  fuse[i] = f[nt * 64 + cc] * mask[nt];
}

__global__ void transpose_kernel(const float* __restrict__ X, float* __restrict__ Y) {
  // (N,T,64) -> (N,64,T). grid (TT/32, 2, NB), block (32,8)
  __shared__ float tile[32][33];
  int n = blockIdx.z, t0 = blockIdx.x * 32, c0 = blockIdx.y * 32;
  for (int r = threadIdx.y; r < 32; r += 8)
    tile[r][threadIdx.x] = X[((long)(n * TT + t0 + r)) * 64 + c0 + threadIdx.x];
  __syncthreads();
  for (int r = threadIdx.y; r < 32; r += 8)
    Y[((long)n * 64 + c0 + r) * TT + t0 + threadIdx.x] = tile[threadIdx.x][r];
}

// ---------------------------------------------------------------------------
// Host orchestration
// ---------------------------------------------------------------------------
extern "C" void kernel_launch(void* const* d_in, const int* in_sizes, int n_in,
                              void* d_out, int out_size, void* d_ws, size_t ws_size,
                              hipStream_t stream) {
  (void)in_sizes; (void)n_in; (void)out_size; (void)ws_size;

  struct Lin  { const float* w; const float* b; };
  struct LinP { const _Float16* w; const float* b; };      // packed weight
  struct SfiP { Lin cs; LinP ff1, ff2; };
  struct LayP { const _Float16* ffw; const float* ffb; LinP q, k, v, o, co; };
  struct PartP { Lin pin; SfiP sfi[9]; LayP lay[10]; LinP cp1[9], cp2[9]; };

  // --- workspace layout (floats)
  const size_t S = (size_t)NB * TT * 64;   // 2,097,152
  float* ws = (float*)d_ws;
  size_t off = 0;
  auto alloc = [&](size_t n) { float* p = ws + off; off += n; return p; };
  float* x10 = alloc((size_t)NB * 10 * TT * 25);
  float* cur[5]; float* spare[5];
  for (int i = 0; i < 5; ++i) { cur[i] = alloc(S); spare[i] = alloc(S); }
  float* t0 = alloc(S); float* t1 = alloc(S); float* t2 = alloc(S);
  float* t3 = alloc(S); float* t4 = alloc(S);          // t0..t4 contiguous = fuse (5S)
  float* amap  = alloc((size_t)NB * 64 * 64);
  float* kvb   = alloc((size_t)NB * NHEAD * 16 * 16);
  float* ksb   = alloc((size_t)NB * NHEAD * 16);
  float* stats = alloc((size_t)NB * 64 * 2);
  off = (off + 15) & ~(size_t)15;                      // 64B-align f16 arena
  _Float16* harena = (_Float16*)(ws + off);
  size_t hoff = 0;
  auto halloc = [&](size_t n) { _Float16* p = harena + hoff; hoff += n; return p; };

  // --- walk d_in in setup_inputs() dict-insertion order; pack weights as we go
  int ix = 0;
  auto nf = [&]() { return (const float*)d_in[ix++]; };
  auto packW = [&](const float* w, int K, int mode) -> const _Float16* {
    _Float16* p = halloc((size_t)64 * K);
    int total = 64 * K;
    pack_kernel<<<dim3((total + 255) / 256), dim3(256), 0, stream>>>(w, p, K, total, mode);
    return p;
  };
  auto linRaw = [&]() { Lin L; L.w = nf(); L.b = nf(); return L; };
  auto linPk  = [&](int K, int mode) { LinP L; L.w = packW(nf(), K, mode); L.b = nf(); return L; };

  const float* X    = nf();
  const float* MASK = nf();
  Lin conv_in = linRaw();
  PartP parts[5];
  for (int i = 0; i < 5; ++i) {
    parts[i].pin = linRaw();
    for (int s = 0; s < 9; ++s) {
      parts[i].sfi[s].cs  = linRaw();
      parts[i].sfi[s].ff1 = linPk(64, 0);              // (in,out)
      parts[i].sfi[s].ff2 = linPk(64, 0);
    }
    for (int l = 0; l < 10; ++l) {
      parts[i].lay[l].ffw = packW(nf(), 192, 2);       // (o,c,3) -> [o][192]
      parts[i].lay[l].ffb = nf();
      parts[i].lay[l].q  = linPk(64, 0);
      parts[i].lay[l].k  = linPk(64, 0);
      parts[i].lay[l].v  = linPk(64, 0);
      parts[i].lay[l].o  = linPk(64, 0);
      parts[i].lay[l].co = linPk(64, 1);               // (out,in) -> straight
    }
    if (i != 4)
      for (int s = 0; s < 9; ++s) { parts[i].cp1[s] = linPk(64, 0); parts[i].cp2[s] = linPk(64, 0); }
  }
  LinP conv_out = linPk(320, 1);                       // (64,320) out-major
  _Float16* amapP = halloc((size_t)NB * 4096);         // packed softmaxed amap

  static const int listOff[5] = {0, 3, 15, 19, 25};
  static const int listLen[5] = {3, 12, 4, 6, 25};

#define GEMM(ACT, MPRE, RES, MPOST, Xp, Wp, wstr, bp, rp, Yp, Kk)                     \
  gemm64_kernel<ACT, MPRE, RES, MPOST><<<dim3(512), dim3(128), 0, stream>>>(          \
      (Xp), (Wp), (wstr), (bp), (rp), MASK, (Yp), (Kk))

  // _cross(fsBuf, feats[i]) -> feats[i]
  auto cross = [&](int i, const float* fsBuf, const LinP& ff1, const LinP& ff2) {
    corr_kernel<<<dim3(NB), dim3(128), 0, stream>>>(fsBuf, cur[i], amap);
    softmax64_kernel<<<dim3(NB), dim3(64), 0, stream>>>(amap);
    pack_kernel<<<dim3(NB * 4096 / 256), dim3(256), 0, stream>>>(amap, amapP, 64, NB * 4096, 1);
    // fc = ft @ amap^T + ft   (per-sample packed weight, stride 4096 halves)
    GEMM(0, false, true, false, cur[i], amapP, 4096, (const float*)nullptr, cur[i], t1, 64);
    GEMM(3, false, false, false, t1, ff1.w, 0, ff1.b, (const float*)nullptr, t2, 64);
    GEMM(0, false, true, true, t2, ff2.w, 0, ff2.b, cur[i], spare[i], 64);
    std::swap(cur[i], spare[i]);
  };

  auto att = [&](int i, const LayP& L, int dil) {
    dilconv_kernel<<<dim3(TT / 64, NB), dim3(128), 0, stream>>>(cur[i], L.ffw, L.ffb, t0, dil);
    instnorm_stats_kernel<<<dim3(NB * 64), dim3(128), 0, stream>>>(t0, stats);
    instnorm_apply_kernel<<<dim3((unsigned)(S / 256)), dim3(256), 0, stream>>>(t0, stats, t1);
    GEMM(2, false, false, false, t1, L.q.w, 0, L.q.b, (const float*)nullptr, t2, 64); // Q=sigmoid
    GEMM(2, false, false, false, t1, L.k.w, 0, L.k.b, (const float*)nullptr, t3, 64); // K=sigmoid
    GEMM(0, false, false, false, t1, L.v.w, 0, L.v.b, (const float*)nullptr, t4, 64); // V
    kv_kernel<<<dim3(NB, NHEAD), dim3(256), 0, stream>>>(t3, t4, kvb, ksb);
    attapply_kernel<<<dim3(NB, TT / 64), dim3(256), 0, stream>>>(t2, kvb, ksb, t1);
    // (att@Wo+bo)*mask + relu_out
    GEMM(0, true, true, false, t1, L.o.w, 0, L.o.b, t0, t2, 64);
    // (x + co(sum)) * mask
    GEMM(0, false, true, true, t2, L.co.w, 0, L.co.b, cur[i], spare[i], 64);
    std::swap(cur[i], spare[i]);
  };

  // ---- pipeline ----
  conv_in_kernel<<<dim3((unsigned)(((size_t)NB * 10 * TT * 25 + 255) / 256)), dim3(256), 0, stream>>>(
      X, conv_in.w, conv_in.b, MASK, x10);

  for (int i = 0; i < 5; ++i)
    nodeconv_kernel<<<dim3((unsigned)(S / 256)), dim3(256), 0, stream>>>(
        x10, parts[i].pin.w, parts[i].pin.b, MASK, cur[i], 9, listOff[i], listLen[i], 1);

  int count = 0;
  for (int index = 0; index < 10; ++index) {
    if (index >= 1) {                       // SFI_LAYER = 1..9
      for (int i = 0; i < 5; ++i) {
        nodeconv_kernel<<<dim3((unsigned)(S / 256)), dim3(256), 0, stream>>>(
            x10, parts[i].sfi[count].cs.w, parts[i].sfi[count].cs.b, MASK, t0,
            count, listOff[i], listLen[i], 0);
        cross(i, t0, parts[i].sfi[count].ff1, parts[i].sfi[count].ff2);
        if (i != 4) cross(i, cur[4], parts[i].cp1[count], parts[i].cp2[count]);
      }
      ++count;
    }
    for (int i = 0; i < 5; ++i) att(i, parts[i].lay[index], 1 << index);
  }

  // fuse (t0..t4 region) + conv_out
  concat_kernel<<<dim3((unsigned)(S * 5 / 256)), dim3(256), 0, stream>>>(
      cur[0], cur[1], cur[2], cur[3], cur[4], MASK, t0);
  GEMM(0, true, false, false, t0, conv_out.w, 0, conv_out.b, (const float*)nullptr, spare[4], 320);

  // outputs: feats[0..3] (N,NF,T) then conv_out (N,64,T)
  float* dout = (float*)d_out;
  for (int i = 0; i < 4; ++i)
    transpose_kernel<<<dim3(TT / 32, 2, NB), dim3(32, 8), 0, stream>>>(cur[i], dout + (size_t)i * S);
  transpose_kernel<<<dim3(TT / 32, 2, NB), dim3(32, 8), 0, stream>>>(spare[4], dout + (size_t)4 * S);

#undef GEMM
}
